// GraphSAGENet_5050881540299
// MI455X (gfx1250) — matrix operand order repaired and checked by reading, not compiled, not run on hardware
//
#include <hip/hip_runtime.h>
#include <hip/hip_bf16.h>

typedef __attribute__((ext_vector_type(16))) __bf16        v16bf;
typedef __attribute__((ext_vector_type(8)))  float         v8f;
typedef __attribute__((ext_vector_type(4)))  unsigned int  v4u;

#define NNODES 50000
#define DFEAT  128
#define DOUT   40

// ---------------------------------------------------------------------------
// Utility kernels
// ---------------------------------------------------------------------------
__global__ void k_zero_f32(float* __restrict__ p, size_t n) {
    size_t i = (size_t)blockIdx.x * blockDim.x + threadIdx.x;
    if (i < n) p[i] = 0.0f;
}

__global__ void k_deg(const long long* __restrict__ dst, float* __restrict__ deg, int E) {
    int e = blockIdx.x * blockDim.x + threadIdx.x;
    if (e < E) atomicAdd(&deg[dst[e]], 1.0f);
}

// 32 lanes per edge, each lane moves 4 contiguous features (float4 gather,
// 4 scalar global_atomic_add_f32 scatters into L2).
__global__ void k_scatter(const float* __restrict__ h,
                          const long long* __restrict__ src,
                          const long long* __restrict__ dst,
                          float* __restrict__ agg, int E) {
    long long t = (long long)blockIdx.x * blockDim.x + threadIdx.x;
    long long e = t >> 5;
    if (e >= E) return;
    int lane = (int)(t & 31);
    long long s = src[e], d = dst[e];
    const float4 v = *(const float4*)(h + s * DFEAT + lane * 4);
    float* ap = agg + d * DFEAT + lane * 4;
    atomicAdd(ap + 0, v.x);
    atomicAdd(ap + 1, v.y);
    atomicAdd(ap + 2, v.z);
    atomicAdd(ap + 3, v.w);
}

__global__ void k_div_deg(float* __restrict__ agg, const float* __restrict__ deg, size_t total) {
    size_t i = (size_t)blockIdx.x * blockDim.x + threadIdx.x;
    if (i >= total) return;
    float dv = fmaxf(deg[i >> 7], 1.0f);
    agg[i] = agg[i] / dv;
}

// WT[b][n][k] = bf16(W[b][k][n])   (row-major K x Nc per batch)
__global__ void k_transpose_bf16(const float* __restrict__ W, __bf16* __restrict__ WT,
                                 int K, int Nc, int batch) {
    long long idx = (long long)blockIdx.x * blockDim.x + threadIdx.x;
    long long per = (long long)K * Nc;
    if (idx >= per * batch) return;
    int b = (int)(idx / per);
    int r = (int)(idx % per);
    int n = r / K, k = r % K;
    WT[(long long)b * per + (long long)n * K + k] = (__bf16)W[(long long)b * per + (long long)k * Nc + n];
}

// W2 is [128 x 40]; pad columns to 48 with zeros, transpose to [48][128] bf16
__global__ void k_w2_pad(const float* __restrict__ W2, __bf16* __restrict__ WT) {
    int idx = blockIdx.x * blockDim.x + threadIdx.x;
    if (idx >= 48 * 128) return;
    int n = idx / 128, k = idx % 128;
    WT[idx] = (n < DOUT) ? (__bf16)W2[k * DOUT + n] : (__bf16)0.0f;
}

// ---------------------------------------------------------------------------
// WMMA fragment loaders (layouts per CDNA5 ISA §7.12.2)
// ---------------------------------------------------------------------------
// A 16x32 bf16 tile from an LDS row-major [16][ldk] bf16 buffer.
// lane L: M = L%16, half = L/16; elems 0..7 -> K = kk*32 + 8*half + e,
//                               elems 8..15 -> K = kk*32 + 8*half + 16 + e.
__device__ __forceinline__ v16bf load_a_frag(const __bf16* base, int ldk, int lane, int kk) {
    int mrow = lane & 15, half = lane >> 4;
    const __bf16* p = base + mrow * ldk + kk * 32 + half * 8;
    union { v16bf v; v4u u[2]; } r;
    r.u[0] = *(const v4u*)(p);
    r.u[1] = *(const v4u*)(p + 16);
    return r.v;
}

// B 32x16 bf16 tile from global WT[n][K] (column n contiguous in K).
// lane L: N = L%16, half = L/16; elems 0..7 -> K = kk*32 + 8*half + e,
//                               elems 8..15 -> K = kk*32 + 8*half + 16 + e.
__device__ __forceinline__ v16bf load_b_frag(const __bf16* __restrict__ WT, int K, int ncol, int half, int kk) {
    const __bf16* p = WT + (long long)ncol * K + kk * 32 + half * 8;
    union { v16bf v; v4u u[2]; } r;
    r.u[0] = *(const v4u*)(p);
    r.u[1] = *(const v4u*)(p + 16);
    return r.v;
}

// ---------------------------------------------------------------------------
// SAGE layer GEMM: out = m @ Wl + h @ Wr + bl     [16-row tile per block]
// 256 threads = 8 waves; wave w owns output N-tile w (16 cols of 128).
// ---------------------------------------------------------------------------
__global__ void k_sage_gemm(const float* __restrict__ m, const float* __restrict__ h,
                            const __bf16* __restrict__ WlT, const __bf16* __restrict__ WrT,
                            const float* __restrict__ bl, float* __restrict__ out) {
    __shared__ __bf16 lm[16 * DFEAT];
    __shared__ __bf16 lh[16 * DFEAT];
    const int tid = threadIdx.x;
    const int row0 = blockIdx.x * 16;
    for (int i = tid; i < 16 * DFEAT; i += 256) {
        int r = i >> 7, c = i & 127;
        size_t gi = (size_t)(row0 + r) * DFEAT + c;
        lm[i] = (__bf16)m[gi];
        lh[i] = (__bf16)h[gi];
    }
    __syncthreads();

    const int lane = tid & 31;
    const int wave = tid >> 5;          // N-tile index 0..7
    const int nloc = lane & 15;
    const int half = lane >> 4;
    const int col  = wave * 16 + nloc;

    v8f acc = {};
#pragma unroll
    for (int kk = 0; kk < 4; ++kk) {
        v16bf aM = load_a_frag(lm, DFEAT, lane, kk);
        v16bf aH = load_a_frag(lh, DFEAT, lane, kk);
        v16bf bL = load_b_frag(WlT, DFEAT, col, half, kk);
        v16bf bR = load_b_frag(WrT, DFEAT, col, half, kk);
        acc = __builtin_amdgcn_wmma_f32_16x16x32_bf16(false, aM, false, bL, (short)0, acc, false, false);
        acc = __builtin_amdgcn_wmma_f32_16x16x32_bf16(false, aH, false, bR, (short)0, acc, false, false);
    }

    const float bias = bl[col];
#pragma unroll
    for (int r = 0; r < 8; ++r) {
        int row = row0 + half * 8 + r;
        out[(size_t)row * DFEAT + col] = acc[r] + bias;
    }
}

// ---------------------------------------------------------------------------
// MLP layer 1: z1 = relu(concat(h1,h2,h3) @ W1 + b1)   K = 384 (12 K-tiles)
// ---------------------------------------------------------------------------
__global__ void k_mlp1_gemm(const float* __restrict__ h1, const float* __restrict__ h2,
                            const float* __restrict__ h3, const __bf16* __restrict__ W1T,
                            const float* __restrict__ b1, float* __restrict__ z1) {
    __shared__ __bf16 lz[16 * 384];
    const int tid = threadIdx.x;
    const int row0 = blockIdx.x * 16;
    for (int i = tid; i < 16 * 384; i += 256) {
        int r = i / 384, c = i % 384;
        const float* src = (c < 128) ? h1 : ((c < 256) ? h2 : h3);
        lz[i] = (__bf16)src[(size_t)(row0 + r) * DFEAT + (c & 127)];
    }
    __syncthreads();

    const int lane = tid & 31;
    const int wave = tid >> 5;
    const int nloc = lane & 15;
    const int half = lane >> 4;
    const int col  = wave * 16 + nloc;

    v8f acc = {};
#pragma unroll
    for (int kk = 0; kk < 12; ++kk) {
        v16bf a = load_a_frag(lz, 384, lane, kk);
        v16bf b = load_b_frag(W1T, 384, col, half, kk);
        acc = __builtin_amdgcn_wmma_f32_16x16x32_bf16(false, a, false, b, (short)0, acc, false, false);
    }

    const float bias = b1[col];
#pragma unroll
    for (int r = 0; r < 8; ++r) {
        int row = row0 + half * 8 + r;
        z1[(size_t)row * DFEAT + col] = fmaxf(acc[r] + bias, 0.0f);
    }
}

// ---------------------------------------------------------------------------
// MLP layer 2: logits = z1 @ W2 + b2   (40 cols padded to 48 -> 3 waves)
// ---------------------------------------------------------------------------
__global__ void k_mlp2_gemm(const float* __restrict__ z1, const __bf16* __restrict__ W2T,
                            const float* __restrict__ b2, float* __restrict__ logits) {
    __shared__ __bf16 lz[16 * DFEAT];
    const int tid = threadIdx.x;     // blockDim = 96 (3 waves)
    const int row0 = blockIdx.x * 16;
    for (int i = tid; i < 16 * DFEAT; i += 96) {
        int r = i >> 7, c = i & 127;
        lz[i] = (__bf16)z1[(size_t)(row0 + r) * DFEAT + c];
    }
    __syncthreads();

    const int lane = tid & 31;
    const int wave = tid >> 5;          // 0..2
    const int nloc = lane & 15;
    const int half = lane >> 4;
    const int col  = wave * 16 + nloc;  // 0..47 (padded)

    v8f acc = {};
#pragma unroll
    for (int kk = 0; kk < 4; ++kk) {
        v16bf a = load_a_frag(lz, DFEAT, lane, kk);
        v16bf b = load_b_frag(W2T, DFEAT, col, half, kk);
        acc = __builtin_amdgcn_wmma_f32_16x16x32_bf16(false, a, false, b, (short)0, acc, false, false);
    }

    if (col < DOUT) {
        const float bias = b2[col];
#pragma unroll
        for (int r = 0; r < 8; ++r) {
            int row = row0 + half * 8 + r;
            logits[(size_t)row * DOUT + col] = acc[r] + bias;
        }
    }
}

// In-place log_softmax over 40 classes, one thread per row.
__global__ void k_logsoftmax(float* __restrict__ out, int n) {
    int row = blockIdx.x * blockDim.x + threadIdx.x;
    if (row >= n) return;
    float* p = out + (size_t)row * DOUT;
    float mx = -INFINITY;
    for (int j = 0; j < DOUT; ++j) mx = fmaxf(mx, p[j]);
    float s = 0.0f;
    for (int j = 0; j < DOUT; ++j) s += __expf(p[j] - mx);
    float ls = __logf(s);
    for (int j = 0; j < DOUT; ++j) p[j] = p[j] - mx - ls;
}

// ---------------------------------------------------------------------------
// Host launcher
// ---------------------------------------------------------------------------
extern "C" void kernel_launch(void* const* d_in, const int* in_sizes, int n_in,
                              void* d_out, int out_size, void* d_ws, size_t ws_size,
                              hipStream_t stream) {
    const int N = NNODES, D = DFEAT;
    const int E = in_sizes[1] / 2;

    const float*     x   = (const float*)d_in[0];
    const long long* ei  = (const long long*)d_in[1];
    const long long* src = ei;
    const long long* dst = ei + E;
    const float* Wl = (const float*)d_in[2];
    const float* Wr = (const float*)d_in[3];
    const float* bl = (const float*)d_in[4];
    const float* W1 = (const float*)d_in[5];
    const float* b1 = (const float*)d_in[6];
    const float* W2 = (const float*)d_in[7];
    const float* b2 = (const float*)d_in[8];

    // --- workspace carve-out (256B aligned slices) ---
    char*  ws  = (char*)d_ws;
    size_t off = 0;
    auto take = [&](size_t bytes) -> char* {
        char* p = ws + off;
        off += (bytes + 255) & ~(size_t)255;
        return p;
    };
    float*  deg = (float*)take((size_t)N * 4);
    float*  agg = (float*)take((size_t)N * D * 4);
    float*  h1  = (float*)take((size_t)N * D * 4);
    float*  h2  = (float*)take((size_t)N * D * 4);
    float*  h3  = (float*)take((size_t)N * D * 4);
    float*  z1  = (float*)take((size_t)N * D * 4);
    __bf16* WlT = (__bf16*)take((size_t)3 * 128 * 128 * 2);
    __bf16* WrT = (__bf16*)take((size_t)3 * 128 * 128 * 2);
    __bf16* W1T = (__bf16*)take((size_t)384 * 128 * 2);
    __bf16* W2T = (__bf16*)take((size_t)48 * 128 * 2);

    const size_t ND = (size_t)N * D;

    // degrees (depend only on dst: compute once)
    k_zero_f32<<<(N + 255) / 256, 256, 0, stream>>>(deg, (size_t)N);
    k_deg<<<(E + 255) / 256, 256, 0, stream>>>(dst, deg, E);

    // weight transposes -> bf16
    k_transpose_bf16<<<(3 * 128 * 128 + 255) / 256, 256, 0, stream>>>(Wl, WlT, 128, 128, 3);
    k_transpose_bf16<<<(3 * 128 * 128 + 255) / 256, 256, 0, stream>>>(Wr, WrT, 128, 128, 3);
    k_transpose_bf16<<<(384 * 128 + 255) / 256, 256, 0, stream>>>(W1, W1T, 384, 128, 1);
    k_w2_pad<<<(48 * 128 + 255) / 256, 256, 0, stream>>>(W2, W2T);

    // SAGE layers
    const float* hprev = x;
    float* houts[3] = {h1, h2, h3};
    for (int l = 0; l < 3; ++l) {
        k_zero_f32<<<(unsigned)((ND + 255) / 256), 256, 0, stream>>>(agg, ND);
        k_scatter<<<(unsigned)(((size_t)E * 32 + 255) / 256), 256, 0, stream>>>(hprev, src, dst, agg, E);
        k_div_deg<<<(unsigned)((ND + 255) / 256), 256, 0, stream>>>(agg, deg, ND);
        k_sage_gemm<<<N / 16, 256, 0, stream>>>(agg, hprev,
                                                WlT + (size_t)l * 128 * 128,
                                                WrT + (size_t)l * 128 * 128,
                                                bl + l * 128, houts[l]);
        hprev = houts[l];
    }

    // MLP head + log_softmax
    k_mlp1_gemm<<<N / 16, 256, 0, stream>>>(h1, h2, h3, W1T, b1, z1);
    k_mlp2_gemm<<<N / 16, 96, 0, stream>>>(z1, W2T, b2, (float*)d_out);
    k_logsoftmax<<<(N + 255) / 256, 256, 0, stream>>>((float*)d_out, N);
}
